// TorchMD_ET_dynamics_32100585570582
// MI455X (gfx1250) — matrix operand order repaired and checked by reading, not compiled, not run on hardware
//
#include <hip/hip_runtime.h>

#define N_NODES 20000
#define N_EDGES 200000
#define HCH     128
#define NT_NODE (N_NODES / 16)   // 1250
#define NT_EDGE (N_EDGES / 16)   // 12500

typedef __attribute__((ext_vector_type(16))) __bf16 bf16x16;
typedef __attribute__((ext_vector_type(8)))  float  floatx8;

// fast silu: x * rcp(1+exp(-x))  (v_exp_f32 + v_rcp_f32, no IEEE div chain)
__device__ __forceinline__ float silu_f(float v) {
    return v * __builtin_amdgcn_rcpf(1.0f + __expf(-v));
}

__device__ __forceinline__ floatx8 wmma_bf16(bf16x16 a, bf16x16 b, floatx8 c) {
    return __builtin_amdgcn_wmma_f32_16x16x32_bf16(false, a, false, b, (short)0, c, false, false);
}

// A tile [16 x 32] from row-major fp32 (base points at row 0, col k0), row stride ld.
// Layout (16-bit A 16x32, matching B): lanes 0-15 -> K 0-15 ; lanes 16-31 -> K 16-31.
__device__ __forceinline__ bf16x16 load_a_f32(const float* base, int ld) {
    int lane = threadIdx.x & 31;
    int m  = lane & 15;
    int kb = (lane >> 4) * 16;         // 0 or 16
    const float* p = base + m * ld + kb;
    bf16x16 a;
#pragma unroll
    for (int e = 0; e < 8; ++e) a[e]     = (__bf16)p[e];
#pragma unroll
    for (int e = 0; e < 8; ++e) a[8 + e] = (__bf16)p[8 + e];
    return a;
}

// B tile [32 x 16] from pre-transposed bf16 weights Wt[N][K] (row stride ldk).
// Layout: lane n = lane%16 ; lanes 0-15 hold K k0..k0+15, lanes 16-31 hold K k0+16..k0+31.
__device__ __forceinline__ bf16x16 load_b_t(const __bf16* wt, int ldk, int k0, int n0) {
    int lane = threadIdx.x & 31;
    const __bf16* p = wt + (size_t)(n0 + (lane & 15)) * ldk + k0 + ((lane >> 4) * 16);
    bf16x16 b;
#pragma unroll
    for (int e = 0; e < 16; ++e) b[e] = p[e];
    return b;
}

// D tile store: lane holds col n = lane%16, rows 8*(lane/16)+r.
__device__ __forceinline__ void store_d_f32(float* base, int ld, floatx8 d) {
    int lane = threadIdx.x & 31;
    int n  = lane & 15;
    int mb = (lane >> 4) * 8;
#pragma unroll
    for (int r = 0; r < 8; ++r) base[(mb + r) * ld + n] = d[r];
}

__device__ __forceinline__ void store_d_bf16(__bf16* base, int ld, floatx8 d) {
    int lane = threadIdx.x & 31;
    int n  = lane & 15;
    int mb = (lane >> 4) * 8;
#pragma unroll
    for (int r = 0; r < 8; ++r) base[(size_t)(mb + r) * ld + n] = (__bf16)d[r];
}

// ---------------- prep kernels ----------------

__global__ void zero_f32_kernel(float* __restrict__ p, int n) {
    int i = blockIdx.x * blockDim.x + threadIdx.x;
    if (i < n) p[i] = 0.0f;
}

__global__ void convert_bf16_kernel(const float* __restrict__ in, __bf16* __restrict__ out, int n) {
    int i = blockIdx.x * blockDim.x + threadIdx.x;
    if (i < n) out[i] = (__bf16)in[i];
}

// w is [K x N] row-major fp32 ; wt is [N x K] row-major bf16.
__global__ void transpose_w_kernel(const float* __restrict__ w, __bf16* __restrict__ wt, int K, int N) {
    int idx = blockIdx.x * blockDim.x + threadIdx.x;
    if (idx < K * N) {
        int n = idx / K, k = idx - n * K;
        wt[idx] = (__bf16)w[k * N + n];
    }
}

// ---------------- node MLP + q/k/v ----------------

__global__ __launch_bounds__(32)
void node_mlp_kernel(const float* __restrict__ x, const float* __restrict__ node_attr,
                     const float* __restrict__ mix_b1, const float* __restrict__ mix_b2,
                     const float* __restrict__ ln_g, const float* __restrict__ ln_b,
                     const float* __restrict__ q_b, const float* __restrict__ k_b,
                     const float* __restrict__ v_b,
                     const __bf16* __restrict__ wt_mix1, const __bf16* __restrict__ wt_mix2,
                     const __bf16* __restrict__ wt_q, const __bf16* __restrict__ wt_k,
                     const __bf16* __restrict__ wt_v,
                     __bf16* __restrict__ q_out, __bf16* __restrict__ k_out,
                     __bf16* __restrict__ v_out) {
    __shared__ float h1[16 * 128];
    __shared__ float h2[16 * 128];
    int tile = blockIdx.x;
    int lane = threadIdx.x;
    const float* xb = x + (size_t)tile * 16 * HCH;
    const float* ab = node_attr + (size_t)tile * 16 * HCH;

    // mix1: [16,256]x[256,128] + silu
    for (int nt = 0; nt < 8; ++nt) {
        floatx8 acc = {};
#pragma unroll
        for (int kk = 0; kk < 4; ++kk)
            acc = wmma_bf16(load_a_f32(xb + kk * 32, HCH), load_b_t(wt_mix1, 256, kk * 32, nt * 16), acc);
#pragma unroll
        for (int kk = 0; kk < 4; ++kk)
            acc = wmma_bf16(load_a_f32(ab + kk * 32, HCH), load_b_t(wt_mix1, 256, 128 + kk * 32, nt * 16), acc);
        float bias = mix_b1[nt * 16 + (lane & 15)];
#pragma unroll
        for (int r = 0; r < 8; ++r) acc[r] = silu_f(acc[r] + bias);
        store_d_f32(h1 + nt * 16, 128, acc);
    }
    __syncthreads();

    // mix2: [16,128]x[128,128]
    for (int nt = 0; nt < 8; ++nt) {
        floatx8 acc = {};
#pragma unroll
        for (int kk = 0; kk < 4; ++kk)
            acc = wmma_bf16(load_a_f32(h1 + kk * 32, 128), load_b_t(wt_mix2, 128, kk * 32, nt * 16), acc);
        float bias = mix_b2[nt * 16 + (lane & 15)];
#pragma unroll
        for (int r = 0; r < 8; ++r) acc[r] += bias;
        store_d_f32(h2 + nt * 16, 128, acc);
    }
    __syncthreads();

    // LayerNorm: 2 lanes per row, 64 channels each; combine halves via shfl_xor(16).
    {
        int m    = lane & 15;
        int half = lane >> 4;
        const float* row = h2 + m * 128 + half * 64;
        float s = 0.0f;
        for (int c = 0; c < 64; ++c) s += row[c];
        s += __shfl_xor(s, 16, 32);
        float mu = s * (1.0f / 128.0f);
        float v2 = 0.0f;
        for (int c = 0; c < 64; ++c) { float d = row[c] - mu; v2 += d * d; }
        v2 += __shfl_xor(v2, 16, 32);
        float inv = rsqrtf(v2 * (1.0f / 128.0f) + 1e-5f);
        for (int c = 0; c < 64; ++c) {
            int cc = half * 64 + c;
            h1[m * 128 + cc] = (row[c] - mu) * inv * ln_g[cc] + ln_b[cc];
        }
    }
    __syncthreads();

    // q, k : [16,128]x[128,128] -> bf16 tables
    for (int nt = 0; nt < 8; ++nt) {
        floatx8 accq = {}, acck = {};
#pragma unroll
        for (int kk = 0; kk < 4; ++kk) {
            bf16x16 a = load_a_f32(h1 + kk * 32, 128);
            accq = wmma_bf16(a, load_b_t(wt_q, 128, kk * 32, nt * 16), accq);
            acck = wmma_bf16(a, load_b_t(wt_k, 128, kk * 32, nt * 16), acck);
        }
        float bq = q_b[nt * 16 + (lane & 15)];
        float bk = k_b[nt * 16 + (lane & 15)];
#pragma unroll
        for (int r = 0; r < 8; ++r) { accq[r] += bq; acck[r] += bk; }
        store_d_bf16(q_out + (size_t)tile * 16 * 128 + nt * 16, 128, accq);
        store_d_bf16(k_out + (size_t)tile * 16 * 128 + nt * 16, 128, acck);
    }
    // v : [16,128]x[128,384]
    for (int nt = 0; nt < 24; ++nt) {
        floatx8 acc = {};
#pragma unroll
        for (int kk = 0; kk < 4; ++kk)
            acc = wmma_bf16(load_a_f32(h1 + kk * 32, 128), load_b_t(wt_v, 128, kk * 32, nt * 16), acc);
        float bias = v_b[nt * 16 + (lane & 15)];
#pragma unroll
        for (int r = 0; r < 8; ++r) acc[r] += bias;
        store_d_bf16(v_out + (size_t)tile * 16 * 384 + nt * 16, 384, acc);
    }
}

// ---------------- vec projection: vp = vec @ vec_w ----------------

__global__ __launch_bounds__(32)
void vec_proj_kernel(const float* __restrict__ vec, const __bf16* __restrict__ wt_vec,
                     float* __restrict__ vec_dot, float* __restrict__ vec3) {
    __shared__ float v1s[16 * 128];
    __shared__ float dot_s[16 * 128];
    int tile = blockIdx.x;
    int lane = threadIdx.x;
    for (int idx = lane; idx < 2048; idx += 32) dot_s[idx] = 0.0f;
    __syncthreads();

    for (int c = 0; c < 3; ++c) {
        const float* vb = vec + (size_t)tile * 16 * 384 + c * 128;   // row stride 384
        // vec1 (cols 0..127)
        for (int nt = 0; nt < 8; ++nt) {
            floatx8 acc = {};
#pragma unroll
            for (int kk = 0; kk < 4; ++kk)
                acc = wmma_bf16(load_a_f32(vb + kk * 32, 384), load_b_t(wt_vec, 128, kk * 32, nt * 16), acc);
            store_d_f32(v1s + nt * 16, 128, acc);
        }
        __syncthreads();
        // vec2 (cols 128..255): accumulate vec1*vec2 into dot_s
        for (int nt = 0; nt < 8; ++nt) {
            floatx8 acc = {};
#pragma unroll
            for (int kk = 0; kk < 4; ++kk)
                acc = wmma_bf16(load_a_f32(vb + kk * 32, 384), load_b_t(wt_vec, 128, kk * 32, 128 + nt * 16), acc);
            int n = lane & 15, mb = (lane >> 4) * 8;
#pragma unroll
            for (int r = 0; r < 8; ++r) {
                int o = (mb + r) * 128 + nt * 16 + n;
                dot_s[o] += v1s[o] * acc[r];
            }
        }
        __syncthreads();
        // vec3 (cols 256..383) -> global
        for (int nt = 0; nt < 8; ++nt) {
            floatx8 acc = {};
#pragma unroll
            for (int kk = 0; kk < 4; ++kk)
                acc = wmma_bf16(load_a_f32(vb + kk * 32, 384), load_b_t(wt_vec, 128, kk * 32, 256 + nt * 16), acc);
            store_d_f32(vec3 + (size_t)tile * 16 * 384 + c * 128 + nt * 16, 384, acc);
        }
        __syncthreads();
    }
    for (int idx = lane; idx < 2048; idx += 32) vec_dot[(size_t)tile * 2048 + idx] = dot_s[idx];
}

// ---------------- fused edge kernel: dk/dv WMMA + attention + messages + scatter ----------------

__global__ __launch_bounds__(32)
void edge_message_kernel(const int* __restrict__ edge_index, const float* __restrict__ r_ij,
                         const float* __restrict__ f_ij, const float* __restrict__ d_ij,
                         const float* __restrict__ dk_b, const float* __restrict__ dv_b,
                         const __bf16* __restrict__ wt_dk, const __bf16* __restrict__ wt_dv,
                         const __bf16* __restrict__ qv, const __bf16* __restrict__ kv,
                         const __bf16* __restrict__ vv, const __bf16* __restrict__ vecbf,
                         float* __restrict__ x_agg, float* __restrict__ vec_agg) {
    __shared__ float dks[16 * 128];
    __shared__ float dvs[16 * 384];
    __shared__ float attn_s[16 * 8];
    int tile = blockIdx.x;
    int lane = threadIdx.x;
    const float* fb = f_ij + (size_t)tile * 16 * 64;

    // dk = silu(f_ij @ dk_w + b) : [16,64]x[64,128]
    for (int nt = 0; nt < 8; ++nt) {
        floatx8 acc = {};
#pragma unroll
        for (int kk = 0; kk < 2; ++kk)
            acc = wmma_bf16(load_a_f32(fb + kk * 32, 64), load_b_t(wt_dk, 64, kk * 32, nt * 16), acc);
        float bias = dk_b[nt * 16 + (lane & 15)];
#pragma unroll
        for (int r = 0; r < 8; ++r) acc[r] = silu_f(acc[r] + bias);
        store_d_f32(dks + nt * 16, 128, acc);
    }
    // dv = silu(f_ij @ dv_w + b) : [16,64]x[64,384]
    for (int nt = 0; nt < 24; ++nt) {
        floatx8 acc = {};
#pragma unroll
        for (int kk = 0; kk < 2; ++kk)
            acc = wmma_bf16(load_a_f32(fb + kk * 32, 64), load_b_t(wt_dv, 64, kk * 32, nt * 16), acc);
        float bias = dv_b[nt * 16 + (lane & 15)];
#pragma unroll
        for (int r = 0; r < 8; ++r) acc[r] = silu_f(acc[r] + bias);
        store_d_f32(dvs + nt * 16, 384, acc);
    }
    __syncthreads();

    // attention: 8 heads x 4 lanes, each lane covers 4 channels of its head
    int h = lane >> 2, p = lane & 3;
    for (int ee = 0; ee < 16; ++ee) {
        int e = tile * 16 + ee;
        int j = edge_index[e];             // source (row 0)
        int i = edge_index[N_EDGES + e];   // destination (row 1)
        float s = 0.0f;
        int c0 = h * 16 + p * 4;
#pragma unroll
        for (int t = 0; t < 4; ++t) {
            int c = c0 + t;
            s += (float)qv[(size_t)i * 128 + c] * (float)kv[(size_t)j * 128 + c] * dks[ee * 128 + c];
        }
        s += __shfl_xor(s, 1, 32);
        s += __shfl_xor(s, 2, 32);
        float r   = r_ij[e];
        float cut = (r < 10.0f) ? 0.5f * (__cosf(r * 0.31415926535f) + 1.0f) : 0.0f;
        float att = silu_f(s) * cut;
        if (p == 0) attn_s[ee * 8 + h] = att;
    }
    __syncthreads();

    // messages + scatter-add
    for (int ee = 0; ee < 16; ++ee) {
        int e = tile * 16 + ee;
        int j = edge_index[e];
        int i = edge_index[N_EDGES + e];
        float d0 = d_ij[e * 3 + 0], d1 = d_ij[e * 3 + 1], d2 = d_ij[e * 3 + 2];
#pragma unroll
        for (int it = 0; it < 4; ++it) {
            int c = it * 32 + lane;
            float att = attn_s[ee * 8 + (c >> 4)];
            float vx = (float)vv[(size_t)j * 384 + c]       * dvs[ee * 384 + c];
            float v1 = (float)vv[(size_t)j * 384 + 128 + c] * dvs[ee * 384 + 128 + c];
            float v2 = (float)vv[(size_t)j * 384 + 256 + c] * dvs[ee * 384 + 256 + c];
            atomicAdd(&x_agg[(size_t)i * 128 + c], vx * att);
            float vb0 = (float)vecbf[(size_t)j * 384 + 0 * 128 + c];
            float vb1 = (float)vecbf[(size_t)j * 384 + 1 * 128 + c];
            float vb2 = (float)vecbf[(size_t)j * 384 + 2 * 128 + c];
            atomicAdd(&vec_agg[(size_t)i * 384 + 0 * 128 + c], vb0 * v1 + v2 * d0);
            atomicAdd(&vec_agg[(size_t)i * 384 + 1 * 128 + c], vb1 * v1 + v2 * d1);
            atomicAdd(&vec_agg[(size_t)i * 384 + 2 * 128 + c], vb2 * v1 + v2 * d2);
        }
    }
}

// ---------------- output kernel: o = x_agg @ o_w ; dx / dvec ----------------

__global__ __launch_bounds__(32)
void output_kernel(const float* __restrict__ x_agg, const __bf16* __restrict__ wt_o,
                   const float* __restrict__ o_b, const float* __restrict__ vec_dot,
                   const float* __restrict__ vec3,
                   float* __restrict__ dx_out, float* __restrict__ dvec_out) {
    __shared__ float os[16 * 384];
    int tile = blockIdx.x;
    int lane = threadIdx.x;
    const float* ab = x_agg + (size_t)tile * 16 * 128;

    for (int nt = 0; nt < 24; ++nt) {
        floatx8 acc = {};
#pragma unroll
        for (int kk = 0; kk < 4; ++kk)
            acc = wmma_bf16(load_a_f32(ab + kk * 32, 128), load_b_t(wt_o, 128, kk * 32, nt * 16), acc);
        float bias = o_b[nt * 16 + (lane & 15)];
#pragma unroll
        for (int r = 0; r < 8; ++r) acc[r] += bias;
        store_d_f32(os + nt * 16, 384, acc);
    }
    __syncthreads();

    // dx = vec_dot * o2 + o3
    for (int idx = lane; idx < 16 * 128; idx += 32) {
        int m = idx >> 7, c = idx & 127;
        size_t g = (size_t)(tile * 16 + m) * 128 + c;
        dx_out[g] = vec_dot[g] * os[m * 384 + 128 + c] + os[m * 384 + 256 + c];
    }
    // dvec = vec3 * o1 + vec_agg (vec_agg already accumulated in dvec_out)
    for (int idx = lane; idx < 16 * 384; idx += 32) {
        int m = idx / 384, rest = idx - m * 384;
        int c = rest & 127;
        size_t g = (size_t)(tile * 16 + m) * 384 + rest;
        dvec_out[g] += vec3[g] * os[m * 384 + c];
    }
}

// ---------------- launch ----------------

extern "C" void kernel_launch(void* const* d_in, const int* in_sizes, int n_in,
                              void* d_out, int out_size, void* d_ws, size_t ws_size,
                              hipStream_t stream) {
    const float* x         = (const float*)d_in[0];
    const float* vec       = (const float*)d_in[1];
    const float* node_attr = (const float*)d_in[2];
    const int*   edge_idx  = (const int*)  d_in[3];
    const float* r_ij      = (const float*)d_in[4];
    const float* f_ij      = (const float*)d_in[5];
    const float* d_ij      = (const float*)d_in[6];
    const float* mix_w1 = (const float*)d_in[7];   const float* mix_b1 = (const float*)d_in[8];
    const float* mix_w2 = (const float*)d_in[9];   const float* mix_b2 = (const float*)d_in[10];
    const float* ln_g   = (const float*)d_in[11];  const float* ln_b   = (const float*)d_in[12];
    const float* q_w    = (const float*)d_in[13];  const float* q_b    = (const float*)d_in[14];
    const float* k_w    = (const float*)d_in[15];  const float* k_b    = (const float*)d_in[16];
    const float* v_w    = (const float*)d_in[17];  const float* v_b    = (const float*)d_in[18];
    const float* o_w    = (const float*)d_in[19];  const float* o_b    = (const float*)d_in[20];
    const float* vec_w  = (const float*)d_in[21];
    const float* dk_w   = (const float*)d_in[22];  const float* dk_b   = (const float*)d_in[23];
    const float* dv_w   = (const float*)d_in[24];  const float* dv_b   = (const float*)d_in[25];

    float* dx_out   = (float*)d_out;                          // [N,128]
    float* dvec_out = dx_out + (size_t)N_NODES * 128;         // [N,3,128]

    // workspace carve-up (256B aligned)
    char*  base = (char*)d_ws;
    size_t off  = 0;
    auto carve = [&](size_t bytes) -> char* {
        char* p = base + off;
        off = (off + bytes + 255) & ~(size_t)255;
        return p;
    };
    __bf16* wt_mix1 = (__bf16*)carve(256 * 128 * 2);
    __bf16* wt_mix2 = (__bf16*)carve(128 * 128 * 2);
    __bf16* wt_q    = (__bf16*)carve(128 * 128 * 2);
    __bf16* wt_k    = (__bf16*)carve(128 * 128 * 2);
    __bf16* wt_v    = (__bf16*)carve(128 * 384 * 2);
    __bf16* wt_o    = (__bf16*)carve(128 * 384 * 2);
    __bf16* wt_vec  = (__bf16*)carve(128 * 384 * 2);
    __bf16* wt_dk   = (__bf16*)carve(64 * 128 * 2);
    __bf16* wt_dv   = (__bf16*)carve(64 * 384 * 2);
    __bf16* q_bf    = (__bf16*)carve((size_t)N_NODES * 128 * 2);
    __bf16* k_bf    = (__bf16*)carve((size_t)N_NODES * 128 * 2);
    __bf16* v_bf    = (__bf16*)carve((size_t)N_NODES * 384 * 2);
    __bf16* vecbf   = (__bf16*)carve((size_t)N_NODES * 384 * 2);
    float*  vdot    = (float*) carve((size_t)N_NODES * 128 * 4);
    float*  vec3    = (float*) carve((size_t)N_NODES * 384 * 4);
    float*  x_agg   = (float*) carve((size_t)N_NODES * 128 * 4);
    (void)ws_size; (void)n_in; (void)in_sizes; (void)out_size;

    const int TPB = 256;
    auto grid1 = [&](int n) { return (n + TPB - 1) / TPB; };

    // prep: zero accumulators (dvec region of d_out collects the atomic segment-sum)
    zero_f32_kernel<<<grid1(N_NODES * 128), TPB, 0, stream>>>(x_agg, N_NODES * 128);
    zero_f32_kernel<<<grid1(N_NODES * 384), TPB, 0, stream>>>(dvec_out, N_NODES * 384);

    // prep: bf16 weight transposes + vec conversion
    transpose_w_kernel<<<grid1(256 * 128), TPB, 0, stream>>>(mix_w1, wt_mix1, 256, 128);
    transpose_w_kernel<<<grid1(128 * 128), TPB, 0, stream>>>(mix_w2, wt_mix2, 128, 128);
    transpose_w_kernel<<<grid1(128 * 128), TPB, 0, stream>>>(q_w,  wt_q,  128, 128);
    transpose_w_kernel<<<grid1(128 * 128), TPB, 0, stream>>>(k_w,  wt_k,  128, 128);
    transpose_w_kernel<<<grid1(128 * 384), TPB, 0, stream>>>(v_w,  wt_v,  128, 384);
    transpose_w_kernel<<<grid1(128 * 384), TPB, 0, stream>>>(o_w,  wt_o,  128, 384);
    transpose_w_kernel<<<grid1(128 * 384), TPB, 0, stream>>>(vec_w, wt_vec, 128, 384);
    transpose_w_kernel<<<grid1(64 * 128),  TPB, 0, stream>>>(dk_w, wt_dk, 64, 128);
    transpose_w_kernel<<<grid1(64 * 384),  TPB, 0, stream>>>(dv_w, wt_dv, 64, 384);
    convert_bf16_kernel<<<grid1(N_NODES * 384), TPB, 0, stream>>>(vec, vecbf, N_NODES * 384);

    // main pipeline
    node_mlp_kernel<<<NT_NODE, 32, 0, stream>>>(x, node_attr, mix_b1, mix_b2, ln_g, ln_b,
                                                q_b, k_b, v_b,
                                                wt_mix1, wt_mix2, wt_q, wt_k, wt_v,
                                                q_bf, k_bf, v_bf);
    vec_proj_kernel<<<NT_NODE, 32, 0, stream>>>(vec, wt_vec, vdot, vec3);
    edge_message_kernel<<<NT_EDGE, 32, 0, stream>>>(edge_idx, r_ij, f_ij, d_ij, dk_b, dv_b,
                                                    wt_dk, wt_dv, q_bf, k_bf, v_bf, vecbf,
                                                    x_agg, dvec_out);
    output_kernel<<<NT_NODE, 32, 0, stream>>>(x_agg, wt_o, o_b, vdot, vec3, dx_out, dvec_out);
}
